// GNN_Encoder_52390011076801
// MI455X (gfx1250) — matrix operand order repaired and checked by reading, not compiled, not run on hardware
//
#include <hip/hip_runtime.h>

#define N_NODES 131072
#define N_EDGES 1048576

typedef __attribute__((ext_vector_type(16))) _Float16 v16h;
typedef __attribute__((ext_vector_type(8)))  float    v8f;

// ---------- order-preserving float<->uint key for atomic max ----------
__device__ __forceinline__ unsigned f32_to_key(float x) {
  unsigned u = __float_as_uint(x);
  return (u & 0x80000000u) ? ~u : (u | 0x80000000u);
}
__device__ __forceinline__ float key_to_f32(unsigned k) {
  unsigned u = (k & 0x80000000u) ? (k ^ 0x80000000u) : ~k;
  return __uint_as_float(u);
}

// ---------- tiny utility kernels ----------
__global__ void k_zero_f32(float* p, int n) {
  int i = blockIdx.x * blockDim.x + threadIdx.x;
  if (i < n) p[i] = 0.0f;
}
__global__ void k_zero_u32(unsigned* p, int n) {
  int i = blockIdx.x * blockDim.x + threadIdx.x;
  if (i < n) p[i] = 0u;
}
__global__ void k_fill_rows(float* __restrict__ out, const float* __restrict__ b,
                            int total, int dout) {
  int i = blockIdx.x * blockDim.x + threadIdx.x;
  if (i < total) out[i] = b[i % dout];
}
__global__ void k_relu(float* p, int n) {
  int i = blockIdx.x * blockDim.x + threadIdx.x;
  if (i < n) p[i] = fmaxf(p[i], 0.0f);
}

// ---------- self-loop edge-attr = mean of incoming attrs ----------
__global__ void k_edge_sums(const int* __restrict__ dst, const float* __restrict__ ea,
                            float* __restrict__ cnt, float* __restrict__ easum,
                            int E, int addCnt) {
  int i = blockIdx.x * blockDim.x + threadIdx.x;
  if (i >= E) return;
  int d = dst[i];
  atomicAdd(&easum[d], ea[i]);
  if (addCnt) atomicAdd(&cnt[d], 1.0f);
}
__global__ void k_finalize_sl(const float* __restrict__ easum, const float* __restrict__ cnt,
                              float* __restrict__ sl, int n) {
  int i = blockIdx.x * blockDim.x + threadIdx.x;
  if (i < n) sl[i] = easum[i] / fmaxf(cnt[i], 1.0f);
}

// ---------- node transform: out[N,DOUT] = x[N,DIN] @ W[DIN,DOUT] + bias, via WMMA ----------
// Branchless zero-padding: full rows/columns are loaded unconditionally (vectorizable),
// padding applied with register selects so no exec-masked load regions are generated.
template <int DIN, int DOUT>
__global__ void k_node_xform(const float* __restrict__ x, const float* __restrict__ W,
                             const float* __restrict__ bias, float* __restrict__ out, int n) {
  static_assert(DIN <= 16, "DIN must fit a single padded K=32 tile");
  const int lane = threadIdx.x & 31;
  const int hf   = lane >> 4;   // half-wave select
  const int r16  = lane & 15;
  const bool hi  = (hf != 0);
  const int wid  = blockIdx.x * (blockDim.x >> 5) + (threadIdx.x >> 5);
  const int nw   = gridDim.x * (blockDim.x >> 5);
  constexpr int CT = (DOUT + 15) / 16;

  // B tiles (32 x 16, K-major per lane): element e -> K = 16*hf + e, N = r16.
  // Since DIN <= 16, only hf==0 lanes carry real weights.
  v16h Bt[CT];
  float bv[CT];
#pragma unroll
  for (int ct = 0; ct < CT; ++ct) {
    const int col   = ct * 16 + r16;
    const bool cok  = (col < DOUT);
    const int colc  = cok ? col : (DOUT - 1);        // clamp -> load always legal
    float wcol[DIN];
#pragma unroll
    for (int k = 0; k < DIN; ++k) wcol[k] = W[k * DOUT + colc];
#pragma unroll
    for (int e = 0; e < 16; ++e) {
      float w = 0.0f;
      if (e < DIN) w = (!hi && cok) ? wcol[e] : 0.0f;  // e<DIN folds at compile time
      Bt[ct][e] = (_Float16)w;
    }
    bv[ct] = cok ? bias[col] : 0.0f;
  }

  const int tiles = n >> 4;
  for (int t = wid; t < tiles; t += nw) {
    const int row = t * 16 + r16;
    float rowv[DIN];
#pragma unroll
    for (int k = 0; k < DIN; ++k) rowv[k] = x[row * DIN + k];  // contiguous -> b96/b128

    // A element e: K = 16*hf + e ; hf==1 -> K >= 16 (always padding, DIN<=16)
    v16h a;
#pragma unroll
    for (int e = 0; e < 16; ++e) {
      float xv = 0.0f;
      if (e < DIN) xv = hi ? 0.0f : rowv[e];
      a[e] = (_Float16)xv;
    }

#pragma unroll
    for (int ct = 0; ct < CT; ++ct) {
      v8f c = {};
      c = __builtin_amdgcn_wmma_f32_16x16x32_f16(false, a, false, Bt[ct],
                                                 (short)0, c, false, false);
      const int col = ct * 16 + r16;
      if (col < DOUT) {
#pragma unroll
        for (int r = 0; r < 8; ++r) {
          const int m = r + 8 * hf;  // D VGPR r: rows r (lanes 0-15), r+8 (lanes 16-31)
          out[(t * 16 + m) * DOUT + col] = c[r] + bv[ct];
        }
      }
    }
  }
}

// ---------- warp-per-edge attention score + segment max ----------
template <int DOUT>
__global__ void k_edge_score(const int* __restrict__ src, const int* __restrict__ dst,
                             const float* __restrict__ ea, const float* __restrict__ sl,
                             const float* __restrict__ xl, const float* __restrict__ xr,
                             const float* __restrict__ We, const float* __restrict__ attv,
                             float* __restrict__ evals, unsigned* __restrict__ mx,
                             int E, int F) {
  int lane = threadIdx.x & 31;
  int f = blockIdx.x * (blockDim.x >> 5) + (threadIdx.x >> 5);
  if (f >= F) return;
  int s = (f < E) ? src[f] : (f - E);
  int d = (f < E) ? dst[f] : (f - E);
  float eav = (f < E) ? ea[f] : sl[f - E];
  float acc = 0.0f;
#pragma unroll
  for (int j0 = 0; j0 < DOUT; j0 += 32) {
    int j = j0 + lane;
    if (j < DOUT) {
      float m = xl[s * DOUT + j] + xr[d * DOUT + j] + eav * We[j];
      float lr = (m > 0.0f) ? m : 0.2f * m;
      acc += lr * attv[j];
    }
  }
  for (int off = 16; off > 0; off >>= 1) acc += __shfl_down(acc, off, 32);
  if (lane == 0) {
    evals[f] = acc;
    atomicMax(&mx[d], f32_to_key(acc));
  }
}

// ---------- exp(e - max) and segment denominator (in place on evals) ----------
__global__ void k_edge_exp(const int* __restrict__ dst, float* __restrict__ ev,
                           const unsigned* __restrict__ mx, float* __restrict__ den,
                           int E, int F) {
  int f = blockIdx.x * blockDim.x + threadIdx.x;
  if (f >= F) return;
  int d = (f < E) ? dst[f] : (f - E);
  float ex = __expf(ev[f] - key_to_f32(mx[d]));
  ev[f] = ex;
  atomicAdd(&den[d], ex);
}

// ---------- warp-per-edge alpha + scatter-add ----------
template <int DOUT>
__global__ void k_alpha_scatter(const int* __restrict__ src, const int* __restrict__ dst,
                                const float* __restrict__ ev, const float* __restrict__ den,
                                const float* __restrict__ xl, float* __restrict__ alpha,
                                float* __restrict__ hout, int E, int F) {
  int lane = threadIdx.x & 31;
  int f = blockIdx.x * (blockDim.x >> 5) + (threadIdx.x >> 5);
  if (f >= F) return;
  int s = (f < E) ? src[f] : (f - E);
  int d = (f < E) ? dst[f] : (f - E);
  float a = ev[f] / den[d];
  if (lane == 0) alpha[f] = a;
#pragma unroll
  for (int j0 = 0; j0 < DOUT; j0 += 32) {
    int j = j0 + lane;
    if (j < DOUT) atomicAdd(&hout[d * DOUT + j], a * xl[s * DOUT + j]);
  }
}

// ---------- output edge index (src|loops, dst|loops) ----------
__global__ void k_write_edges(const int* __restrict__ src, const int* __restrict__ dst,
                              int* __restrict__ fs, int* __restrict__ fd, int E, int F) {
  int f = blockIdx.x * blockDim.x + threadIdx.x;
  if (f >= F) return;
  fs[f] = (f < E) ? src[f] : (f - E);
  fd[f] = (f < E) ? dst[f] : (f - E);
}

// ---------- one GATv2 layer ----------
template <int DIN, int DOUT>
static void run_layer(const float* hin, const int* src, const int* dst, const float* eaPtr,
                      const float* Wl, const float* bl, const float* Wr, const float* br,
                      const float* We, const float* attv, const float* b,
                      float* hout, float* alphaOut, bool relu, bool computeCnt,
                      float* xl, float* xr, float* ev, float* cnt, float* easum,
                      float* sl, unsigned* mx, float* den, hipStream_t stream) {
  const int Nn = N_NODES, E = N_EDGES, F = E + Nn;
  const int nb = (Nn + 255) / 256;
  k_zero_f32<<<nb, 256, 0, stream>>>(easum, Nn);
  k_zero_u32<<<nb, 256, 0, stream>>>(mx, Nn);
  k_zero_f32<<<nb, 256, 0, stream>>>(den, Nn);
  if (computeCnt) k_zero_f32<<<nb, 256, 0, stream>>>(cnt, Nn);
  k_edge_sums<<<(E + 255) / 256, 256, 0, stream>>>(dst, eaPtr, cnt, easum, E,
                                                   computeCnt ? 1 : 0);
  k_finalize_sl<<<nb, 256, 0, stream>>>(easum, cnt, sl, Nn);
  // node transforms: 8192 tiles of 16 nodes, 8 waves/block -> 1024 blocks
  k_node_xform<DIN, DOUT><<<1024, 256, 0, stream>>>(hin, Wl, bl, xl, Nn);
  k_node_xform<DIN, DOUT><<<1024, 256, 0, stream>>>(hin, Wr, br, xr, Nn);
  const int wb = (F + 7) / 8;   // warp-per-edge, 8 waves per 256-thread block
  k_edge_score<DOUT><<<wb, 256, 0, stream>>>(src, dst, eaPtr, sl, xl, xr, We, attv,
                                             ev, mx, E, F);
  k_edge_exp<<<(F + 255) / 256, 256, 0, stream>>>(dst, ev, mx, den, E, F);
  k_fill_rows<<<((size_t)Nn * DOUT + 255) / 256, 256, 0, stream>>>(hout, b, Nn * DOUT, DOUT);
  k_alpha_scatter<DOUT><<<wb, 256, 0, stream>>>(src, dst, ev, den, xl, alphaOut, hout, E, F);
  if (relu) k_relu<<<((size_t)Nn * DOUT + 255) / 256, 256, 0, stream>>>(hout, Nn * DOUT);
}

extern "C" void kernel_launch(void* const* d_in, const int* in_sizes, int n_in,
                              void* d_out, int out_size, void* d_ws, size_t ws_size,
                              hipStream_t stream) {
  (void)in_sizes; (void)n_in; (void)out_size; (void)ws_size;
  const int Nn = N_NODES, E = N_EDGES, F = E + Nn;

  const float* x     = (const float*)d_in[0];
  const int*   ei    = (const int*)d_in[1];
  const float* eattr = (const float*)d_in[2];
  const float* P[21];
  for (int i = 0; i < 21; ++i) P[i] = (const float*)d_in[3 + i];
  const int* src = ei;
  const int* dst = ei + E;

  float* wsf = (float*)d_ws;
  size_t off = 0;
  float* xl    = wsf + off; off += (size_t)Nn * 64;
  float* xr    = wsf + off; off += (size_t)Nn * 64;
  float* h1    = wsf + off; off += (size_t)Nn * 8;
  float* h2    = wsf + off; off += (size_t)Nn * 16;
  float* ev    = wsf + off; off += (size_t)F;
  float* a1    = wsf + off; off += (size_t)F;
  float* a2    = wsf + off; off += (size_t)F;
  float* cnt   = wsf + off; off += (size_t)Nn;
  float* easum = wsf + off; off += (size_t)Nn;
  float* sl    = wsf + off; off += (size_t)Nn;
  float* den   = wsf + off; off += (size_t)Nn;
  unsigned* mx = (unsigned*)(wsf + off); off += (size_t)Nn;

  float* hfin = (float*)d_out;                                   // N x 64
  int*   fs   = (int*)d_out + (size_t)Nn * 64;                   // F ints
  int*   fd   = fs + F;                                          // F ints
  float* a3   = (float*)d_out + (size_t)Nn * 64 + 2 * (size_t)F; // F floats

  // layer 1: 3 -> 8, ea = edge_attr
  run_layer<3, 8>(x, src, dst, eattr, P[0], P[1], P[2], P[3], P[4], P[5], P[6],
                  h1, a1, true, true, xl, xr, ev, cnt, easum, sl, mx, den, stream);
  // layer 2: 8 -> 16, ea = a1[:E]
  run_layer<8, 16>(h1, src, dst, a1, P[7], P[8], P[9], P[10], P[11], P[12], P[13],
                   h2, a2, true, false, xl, xr, ev, cnt, easum, sl, mx, den, stream);
  // layer 3: 16 -> 64, ea = a2[:E]; h and alpha written straight into d_out
  run_layer<16, 64>(h2, src, dst, a2, P[14], P[15], P[16], P[17], P[18], P[19], P[20],
                    hfin, a3, false, false, xl, xr, ev, cnt, easum, sl, mx, den, stream);

  k_write_edges<<<(F + 255) / 256, 256, 0, stream>>>(src, dst, fs, fd, E, F);
}
